// GPNNAStar_85126251807256
// MI455X (gfx1250) — matrix operand order repaired and checked by reading, not compiled, not run on hardware
//
#include <hip/hip_runtime.h>
#include <math.h>

#define NREG 128
#define BATCH 512
#define DDIM 128
#define TSTEPS 16
#define NEGINF -1e9f

typedef __attribute__((ext_vector_type(16))) _Float16 v16h;
typedef __attribute__((ext_vector_type(8)))  float    v8f;

enum { EPI_PLAIN = 0, EPI_RELU = 1, EPI_LAYER = 2, EPI_PROJ = 3 };

// Y[M,J] = X[M,K] @ W[J,K]^T + bias, with per-mode epilogue.
// One wave computes one 16x16 output tile via v_wmma_f32_16x16x32_f16.
template<int EPI>
__global__ void wmma_gemm(const float* __restrict__ X, int ldx,
                          const float* __restrict__ W,
                          const float* __restrict__ bias,
                          float* __restrict__ Y,
                          int M, int Jdim, int K,
                          const float* __restrict__ rscale,
                          const float* __restrict__ agg)
{
  const int gwave = (int)((blockIdx.x * blockDim.x + threadIdx.x) >> 5);
  const int lane  = (int)(threadIdx.x & 31);
  const int jt = Jdim >> 4;
  const int mtile = gwave / jt;
  const int j0 = (gwave - mtile * jt) << 4;
  const int m0 = mtile << 4;
  if (m0 >= M) return;
  const int half = lane >> 4;   // 0: lanes 0-15, 1: lanes 16-31
  const int l16  = lane & 15;

  const float* xrow = X + (size_t)(m0 + l16) * (size_t)ldx;       // A row m0+l16
  const float* wrow = W + (size_t)(j0 + l16) * (size_t)K;         // B col j0+l16

  v8f acc = {0.f, 0.f, 0.f, 0.f, 0.f, 0.f, 0.f, 0.f};
  for (int kb = 0; kb < K; kb += 32) {
    v16h a, b;
    // A 16x32 f16 layout: lanes<16: K = kb+0..7 (v16h[0..7]) and kb+16..23 (v16h[8..15]);
    //                     lanes>=16: K = kb+8..15 and kb+24..31.
    const int ka = kb + half * 8;
    #pragma unroll
    for (int i = 0; i < 8; ++i) a[i]     = (_Float16)xrow[ka + i];
    #pragma unroll
    for (int i = 0; i < 8; ++i) a[8 + i] = (_Float16)xrow[ka + 16 + i];
    // B 32x16 f16 layout: same K order as A: lanes<16: K = kb+0..7 and kb+16..23; lanes>=16: kb+8..15 and kb+24..31.
    const int kw = kb + half * 8;
    #pragma unroll
    for (int i = 0; i < 16; ++i) b[i] = (_Float16)wrow[kw + (i & 7) + ((i >> 3) << 4)];
    acc = __builtin_amdgcn_wmma_f32_16x16x32_f16(false, a, false, b,
                                                 (short)0, acc, false, false);
  }

  const int col = j0 + l16;
  const float bj = bias[col];
  float rr = 0.f, omr = 0.f;
  if (EPI == EPI_LAYER) { rr = rscale[0]; omr = 1.f - rr; }
  // C/D layout: VGPR r holds (row m0 + half*8 + r, col j0 + lane%16)
  #pragma unroll
  for (int r = 0; r < 8; ++r) {
    const int row = m0 + half * 8 + r;
    float v = acc[r] + bj;
    if (EPI == EPI_RELU) v = fmaxf(v, 0.f);
    size_t oidx;
    if (EPI == EPI_PROJ) {              // rows are m = n*BATCH + b ; store e[b][n][col]
      const int n  = row >> 9;          // / BATCH
      const int bb = row & (BATCH - 1);
      oidx = ((size_t)bb * NREG + n) * DDIM + col;
    } else if (EPI == EPI_LAYER) {      // r*lin + (1-r)*relu(agg(mean)) broadcast per b
      const int bb = row & (BATCH - 1);
      v = rr * v + omr * agg[bb * DDIM + col];
      oidx = (size_t)row * (size_t)Jdim + col;
    } else {
      oidx = (size_t)row * (size_t)Jdim + col;
    }
    Y[oidx] = v;
  }
}

// emb[n,b,:] = data[n,b,0:4] @ node_embed[4,D]; one block (128 thr) per (n,b)
__global__ void embed_kernel(const float* __restrict__ data,
                             const float* __restrict__ node_embed,
                             float* __restrict__ emb)
{
  const int nb = blockIdx.x;
  const int d  = threadIdx.x;
  const float f0 = data[nb * 4 + 0], f1 = data[nb * 4 + 1];
  const float f2 = data[nb * 4 + 2], f3 = data[nb * 4 + 3];
  emb[(size_t)nb * DDIM + d] =
      f0 * node_embed[0 * DDIM + d] + f1 * node_embed[1 * DDIM + d] +
      f2 * node_embed[2 * DDIM + d] + f3 * node_embed[3 * DDIM + d];
}

// mean over n of src[N,B,D] -> mean[B,D]
__global__ void mean_over_n(const float* __restrict__ src, float* __restrict__ mean)
{
  const int t = blockIdx.x * blockDim.x + threadIdx.x;   // b*D + d
  float s = 0.f;
  for (int n = 0; n < NREG; ++n) s += src[(size_t)n * (BATCH * DDIM) + t];
  mean[t] = s * (1.f / (float)NREG);
}

// Wcat[j, 0:128]=Wih[j], [128:256]=Whh[j]; bcat = bih + bhh
__global__ void build_lstm_cat(const float* __restrict__ Wih, const float* __restrict__ Whh,
                               const float* __restrict__ bih, const float* __restrict__ bhh,
                               float* __restrict__ Wcat, float* __restrict__ bcat)
{
  const int t = blockIdx.x * blockDim.x + threadIdx.x;   // 512*256
  const int j = t >> 8, k = t & 255;
  Wcat[t] = (k < DDIM) ? Wih[j * DDIM + k] : Whh[j * DDIM + (k - DDIM)];
  if (k == 0) bcat[j] = bih[j] + bhh[j];
}

// xh[b, 0:128] = init_placeholder; xh[b,128:256]=0 (h0); c=0; visited=0
__global__ void init_state(const float* __restrict__ initp, float* __restrict__ xh,
                           float* __restrict__ c, unsigned* __restrict__ visited)
{
  const int t = blockIdx.x * blockDim.x + threadIdx.x;   // B*256
  const int b = t >> 8, k = t & 255;
  xh[t] = (k < DDIM) ? initp[k] : 0.f;
  if (k < DDIM) c[b * DDIM + k] = 0.f;
  if (k < 4) visited[b * 4 + k] = 0u;
}

__device__ __forceinline__ float sigm(float x) { return 1.f / (1.f + __expf(-x)); }

// gates[b,512] (i,f,g,o) -> update c, write h into xh[b,128:256]
__global__ void lstm_update(const float* __restrict__ gates,
                            float* __restrict__ c, float* __restrict__ xh)
{
  const int t = blockIdx.x * blockDim.x + threadIdx.x;   // B*D
  const int b = t >> 7, d = t & 127;
  const float* g = gates + (size_t)b * 512;
  const float gi = sigm(g[d]);
  const float gf = sigm(g[DDIM + d]);
  const float gg = tanhf(g[2 * DDIM + d]);
  const float go = sigm(g[3 * DDIM + d]);
  const float cn = gf * c[t] + gi * gg;
  c[t] = cn;
  xh[b * 256 + DDIM + d] = go * tanhf(cn);
}

// lg[b,n] = mask( clip * v . tanh(qq[b,:] + e[b,n,:]) ); one wave per (b,n)
__global__ void attn_logits(const float* __restrict__ qq, const float* __restrict__ e,
                            const float* __restrict__ v, const unsigned* __restrict__ visited,
                            float* __restrict__ lg, float clipC)
{
  __shared__ float sqq[DDIM];
  __shared__ float sv[DDIM];
  const int b = blockIdx.x >> 4;
  const int nbase = (blockIdx.x & 15) << 3;
  const int t = threadIdx.x;
  if (t < DDIM) sqq[t] = qq[b * DDIM + t];
  else sv[t - DDIM] = v[t - DDIM];
  __syncthreads();
  const int wave = t >> 5, lane = t & 31;
  const int n = nbase + wave;
  const float* er = e + ((size_t)b * NREG + n) * DDIM;
  float s = 0.f;
  #pragma unroll
  for (int k = lane; k < DDIM; k += 32) s += sv[k] * tanhf(sqq[k] + er[k]);
  #pragma unroll
  for (int off = 16; off > 0; off >>= 1) s += __shfl_down(s, off, 32);
  if (lane == 0) {
    if (clipC > 0.f) s = clipC * tanhf(s);
    const bool vis = (visited[b * 4 + (n >> 5)] >> (n & 31)) & 1u;
    lg[b * NREG + n] = vis ? NEGINF : s;
  }
}

// per b: w = softmax(lg[b,:]); gl[b,d] = sum_n w[n] * e[b,n,d]
__global__ void glimpse_kernel(const float* __restrict__ lg, const float* __restrict__ e,
                               float* __restrict__ gl)
{
  __shared__ float red[NREG];
  __shared__ float w[NREG];
  const int b = blockIdx.x, t = threadIdx.x;
  const float x = lg[b * NREG + t];
  red[t] = x; __syncthreads();
  for (int s = 64; s > 0; s >>= 1) { if (t < s) red[t] = fmaxf(red[t], red[t + s]); __syncthreads(); }
  const float m = red[0]; __syncthreads();
  const float ex = __expf(x - m);
  red[t] = ex; __syncthreads();
  for (int s = 64; s > 0; s >>= 1) { if (t < s) red[t] += red[t + s]; __syncthreads(); }
  w[t] = ex / red[0]; __syncthreads();
  float acc = 0.f;
  const float* eb = e + (size_t)b * NREG * DDIM;
  for (int n = 0; n < NREG; ++n) acc += w[n] * eb[n * DDIM + t];
  gl[b * DDIM + t] = acc;
}

// per b: log_softmax(lp) -> out; argmax (first) -> sel; visited |= sel; xh[b,0:128] = emb[sel,b,:]
__global__ void select_store(const float* __restrict__ lp, const float* __restrict__ emb,
                             float* __restrict__ out_t, unsigned* __restrict__ visited,
                             float* __restrict__ xh)
{
  __shared__ float red[NREG];
  __shared__ int  redi[NREG];
  __shared__ int  ssel;
  const int b = blockIdx.x, t = threadIdx.x;
  const float x = lp[b * NREG + t];
  red[t] = x; __syncthreads();
  for (int s = 64; s > 0; s >>= 1) { if (t < s) red[t] = fmaxf(red[t], red[t + s]); __syncthreads(); }
  const float m = red[0]; __syncthreads();
  red[t] = __expf(x - m); __syncthreads();
  for (int s = 64; s > 0; s >>= 1) { if (t < s) red[t] += red[t + s]; __syncthreads(); }
  const float lse = m + __logf(red[0]);
  out_t[b * NREG + t] = x - lse;
  __syncthreads();
  red[t] = x; redi[t] = t; __syncthreads();
  for (int s = 64; s > 0; s >>= 1) {
    if (t < s) {
      const float a = red[t], bb = red[t + s];
      if (bb > a || (bb == a && redi[t + s] < redi[t])) { red[t] = bb; redi[t] = redi[t + s]; }
    }
    __syncthreads();
  }
  if (t == 0) {
    const int sel = redi[0];
    ssel = sel;
    visited[b * 4 + (sel >> 5)] |= (1u << (sel & 31));
  }
  __syncthreads();
  const int sel = ssel;
  xh[b * 256 + t] = emb[(size_t)sel * (BATCH * DDIM) + b * DDIM + t];
}

static inline int gemm_blocks(int M, int J) {
  const int waves = (M / 16) * (J / 16);
  return (waves + 7) / 8;
}

extern "C" void kernel_launch(void* const* d_in, const int* in_sizes, int n_in,
                              void* d_out, int out_size, void* d_ws, size_t ws_size,
                              hipStream_t stream) {
  (void)in_sizes; (void)n_in; (void)out_size; (void)ws_size;
  const float* data       = (const float*)d_in[0];
  const float* node_embed = (const float*)d_in[1];
  const float* initp      = (const float*)d_in[2];
  const float* W1_w = (const float*)d_in[3];  const float* W1_b = (const float*)d_in[4];
  const float* W2_w = (const float*)d_in[5];  const float* W2_b = (const float*)d_in[6];
  const float* W3_w = (const float*)d_in[7];  const float* W3_b = (const float*)d_in[8];
  const float* ag1w = (const float*)d_in[9];  const float* ag1b = (const float*)d_in[10];
  const float* ag2w = (const float*)d_in[11]; const float* ag2b = (const float*)d_in[12];
  const float* ag3w = (const float*)d_in[13]; const float* ag3b = (const float*)d_in[14];
  const float* r1 = (const float*)d_in[15];
  const float* r2 = (const float*)d_in[16];
  const float* r3 = (const float*)d_in[17];
  const float* Wih = (const float*)d_in[18]; const float* Whh = (const float*)d_in[19];
  const float* bih = (const float*)d_in[20]; const float* bhh = (const float*)d_in[21];
  const float* ptr_Wq  = (const float*)d_in[22]; const float* ptr_bq  = (const float*)d_in[23];
  const float* ptr_Wref= (const float*)d_in[24]; const float* ptr_bref= (const float*)d_in[25];
  const float* ptr_v   = (const float*)d_in[26];
  const float* glm_Wq  = (const float*)d_in[27]; const float* glm_bq  = (const float*)d_in[28];
  const float* glm_Wref= (const float*)d_in[29]; const float* glm_bref= (const float*)d_in[30];
  const float* glm_v   = (const float*)d_in[31];

  float* ws = (float*)d_ws;
  const size_t BIG = (size_t)NREG * BATCH * DDIM;          // 8388608 floats
  float* emb   = ws; ws += BIG;
  float* ctxA  = ws; ws += BIG;
  float* ctxB  = ws; ws += BIG;
  float* e_glm = ws; ws += BIG;
  float* e_ptr = ws; ws += BIG;
  float* meanb = ws; ws += BATCH * DDIM;
  float* aggR  = ws; ws += BATCH * DDIM;
  float* Wcat  = ws; ws += 512 * 256;
  float* bcat  = ws; ws += 512;
  float* xh    = ws; ws += BATCH * 256;
  float* cbuf  = ws; ws += BATCH * DDIM;
  float* gates = ws; ws += BATCH * 512;
  float* qq    = ws; ws += BATCH * DDIM;
  float* lg    = ws; ws += BATCH * NREG;
  float* gl    = ws; ws += BATCH * DDIM;
  unsigned* visited = (unsigned*)ws;
  float* out = (float*)d_out;

  const int M = NREG * BATCH;   // 65536

  // ---- encoder ----
  embed_kernel<<<NREG * BATCH, DDIM, 0, stream>>>(data, node_embed, emb);
  build_lstm_cat<<<(512 * 256) / 256, 256, 0, stream>>>(Wih, Whh, bih, bhh, Wcat, bcat);
  init_state<<<(BATCH * 256) / 256, 256, 0, stream>>>(initp, xh, cbuf, visited);

  // layer 1: emb -> ctxA
  mean_over_n<<<(BATCH * DDIM) / 256, 256, 0, stream>>>(emb, meanb);
  wmma_gemm<EPI_RELU ><<<gemm_blocks(BATCH, DDIM), 256, 0, stream>>>(meanb, DDIM, ag1w, ag1b, aggR, BATCH, DDIM, DDIM, nullptr, nullptr);
  wmma_gemm<EPI_LAYER><<<gemm_blocks(M, DDIM), 256, 0, stream>>>(emb, DDIM, W1_w, W1_b, ctxA, M, DDIM, DDIM, r1, aggR);
  // layer 2: ctxA -> ctxB
  mean_over_n<<<(BATCH * DDIM) / 256, 256, 0, stream>>>(ctxA, meanb);
  wmma_gemm<EPI_RELU ><<<gemm_blocks(BATCH, DDIM), 256, 0, stream>>>(meanb, DDIM, ag2w, ag2b, aggR, BATCH, DDIM, DDIM, nullptr, nullptr);
  wmma_gemm<EPI_LAYER><<<gemm_blocks(M, DDIM), 256, 0, stream>>>(ctxA, DDIM, W2_w, W2_b, ctxB, M, DDIM, DDIM, r2, aggR);
  // layer 3: ctxB -> ctxA (final ctx)
  mean_over_n<<<(BATCH * DDIM) / 256, 256, 0, stream>>>(ctxB, meanb);
  wmma_gemm<EPI_RELU ><<<gemm_blocks(BATCH, DDIM), 256, 0, stream>>>(meanb, DDIM, ag3w, ag3b, aggR, BATCH, DDIM, DDIM, nullptr, nullptr);
  wmma_gemm<EPI_LAYER><<<gemm_blocks(M, DDIM), 256, 0, stream>>>(ctxB, DDIM, W3_w, W3_b, ctxA, M, DDIM, DDIM, r3, aggR);
  // ref projections (fixed over decode): e[b,n,d]
  wmma_gemm<EPI_PROJ><<<gemm_blocks(M, DDIM), 256, 0, stream>>>(ctxA, DDIM, glm_Wref, glm_bref, e_glm, M, DDIM, DDIM, nullptr, nullptr);
  wmma_gemm<EPI_PROJ><<<gemm_blocks(M, DDIM), 256, 0, stream>>>(ctxA, DDIM, ptr_Wref, ptr_bref, e_ptr, M, DDIM, DDIM, nullptr, nullptr);

  // ---- decode: T sequential steps ----
  for (int t = 0; t < TSTEPS; ++t) {
    // LSTM gates = [x,h] @ [Wih|Whh]^T + (bih+bhh)
    wmma_gemm<EPI_PLAIN><<<gemm_blocks(BATCH, 512), 256, 0, stream>>>(xh, 256, Wcat, bcat, gates, BATCH, 512, 256, nullptr, nullptr);
    lstm_update<<<(BATCH * DDIM) / 256, 256, 0, stream>>>(gates, cbuf, xh);
    // glimpse attention
    wmma_gemm<EPI_PLAIN><<<gemm_blocks(BATCH, DDIM), 256, 0, stream>>>(xh + DDIM, 256, glm_Wq, glm_bq, qq, BATCH, DDIM, DDIM, nullptr, nullptr);
    attn_logits<<<BATCH * (NREG / 8), 256, 0, stream>>>(qq, e_glm, glm_v, visited, lg, 0.f);
    glimpse_kernel<<<BATCH, NREG, 0, stream>>>(lg, e_glm, gl);
    // pointer attention (tanh clipping C=10)
    wmma_gemm<EPI_PLAIN><<<gemm_blocks(BATCH, DDIM), 256, 0, stream>>>(gl, DDIM, ptr_Wq, ptr_bq, qq, BATCH, DDIM, DDIM, nullptr, nullptr);
    attn_logits<<<BATCH * (NREG / 8), 256, 0, stream>>>(qq, e_ptr, ptr_v, visited, lg, 10.f);
    select_store<<<BATCH, NREG, 0, stream>>>(lg, emb, out + (size_t)t * BATCH * NREG, visited, xh);
  }
}
